// OccupancyEncoder_3667902071058
// MI455X (gfx1250) — matrix operand order repaired and checked by reading, not compiled, not run on hardware
//
#include <hip/hip_runtime.h>

typedef __attribute__((ext_vector_type(16))) _Float16 v16h;
typedef __attribute__((ext_vector_type(8)))  float    v8f;

#define Z_DIM 32
#define Y_DIM 256
#define X_DIM 256

// ---------------------------------------------------------------------------
// Scatter input voxels: fp32 [N,16] + coors[x,y,z] -> dense f16 [Z,Y,X,16]
// ---------------------------------------------------------------------------
__global__ __launch_bounds__(256) void k_scatter(
    const float* __restrict__ vf, const int* __restrict__ coors,
    _Float16* __restrict__ act, int N)
{
  int v = blockIdx.x * 256 + threadIdx.x;
  if (v >= N) return;
  int cx = coors[3 * v + 0];
  int cy = coors[3 * v + 1];
  int cz = coors[3 * v + 2];
  size_t dst = ((((size_t)cz * Y_DIM) + cy) * X_DIM + cx) * 16;
  const float* s = vf + (size_t)v * 16;
#pragma unroll
  for (int i = 0; i < 16; ++i) act[dst + i] = (_Float16)s[i];
}

// ---------------------------------------------------------------------------
// Pack conv weights [27*Cin, Cout] fp32 -> f16 B-fragments, CHUNK-major:
//   idx = ((kc*Nt + nt)*32 + lane)*16 + e
//   lane<16  : column n = nt*16+lane,    K = kc*32 + (e&7) + (e>>3)*16      (e = 0..15)
//   lane>=16 : column n = nt*16+lane-16, K = kc*32 + 8 + (e&7) + (e>>3)*16
// K beyond Ktot (first layer: 432 -> padded to 448) packs zeros.
// ---------------------------------------------------------------------------
__global__ __launch_bounds__(256) void k_pack(
    const float* __restrict__ w, _Float16* __restrict__ pB,
    int Cout, int Ktot, int Nt, long long total)
{
  long long idx = (long long)blockIdx.x * 256 + threadIdx.x;
  if (idx >= total) return;
  int e    = (int)(idx & 15);
  int lane = (int)((idx >> 4) & 31);
  long long chunk = idx >> 9;
  int nt = (int)(chunk % Nt);
  int kc = (int)(chunk / Nt);
  int n = nt * 16 + (lane & 15);
  int k = kc * 32 + ((lane >> 4) << 3) + (e & 7) + ((e >> 3) << 4);
  pB[idx] = (k < Ktot) ? (_Float16)w[(size_t)k * Cout + n] : (_Float16)0.f;
}

// ---------------------------------------------------------------------------
// Implicit-GEMM 3x3x3 conv, pad=1, stride 1 or 2, via v_wmma_f32_16x16x32_f16.
// 2D grid: blockIdx.x -> group of 8*MTU m-tiles (MTU per wave), blockIdx.y ->
// group of NTU n-tiles. One wave computes (16*MTU voxels) x (16*NTU channels):
// per 32-wide K chunk it issues 2*MTU A-loads + 2*NTU B-loads for MTU*NTU
// WMMAs, sharing the tap decode across m-tiles and each B fragment across
// m-tiles. No divergence anywhere: EXEC stays all-ones for WMMA.
//   A per-lane: two contiguous 8-half runs within one conv tap -> 2x b128.
//   B per-lane: NTU x 32 contiguous bytes of the packed fragment stream.
// ---------------------------------------------------------------------------
template <int MTU, int NTU>
__global__ __launch_bounds__(256) void k_conv_wmma(
    const _Float16* __restrict__ act, const _Float16* __restrict__ pB,
    const float* __restrict__ bias, float* __restrict__ out,
    int Din, int Hin, int Win, int cin_shift,
    int Hout_mask, int Wout_mask, int ho_shift, int wo_shift,
    int Cout, int stride, int Nt, int KC)
{
  int wave = threadIdx.x >> 5;
  int lane = threadIdx.x & 31;
  int mtbase = (blockIdx.x * 8 + wave) * MTU;
  int ntbase = blockIdx.y * NTU;

  int hs = lane >> 4;     // lane half-select
  int lr = lane & 15;

  // A-matrix rows (voxels) handled by this lane, one per m-sub-tile
  int ox[MTU], oy[MTU], oz[MTU];
#pragma unroll
  for (int i = 0; i < MTU; ++i) {
    int ma = (mtbase + i) * 16 + lr;
    ox[i] = ma & Wout_mask;
    oy[i] = (ma >> wo_shift) & Hout_mask;
    oz[i] = ma >> (wo_shift + ho_shift);
  }
  int Cin_mask = (1 << cin_shift) - 1;

  v8f acc[MTU][NTU];
#pragma unroll
  for (int i = 0; i < MTU; ++i)
#pragma unroll
    for (int j = 0; j < NTU; ++j)
      acc[i][j] = (v8f){0.f, 0.f, 0.f, 0.f, 0.f, 0.f, 0.f, 0.f};

  // chunk-major packed weights: lane's 32B for (kc, nt) at
  //   pB + ((kc*Nt + nt)*512) + lane*16
  const _Float16* pBl = pB + (size_t)ntbase * 512 + (size_t)lane * 16;
  size_t chunk_stride = (size_t)Nt * 512;

  for (int kc = 0; kc < KC; ++kc) {
    // ---- A fragments: two 8-half runs (im2col); tap decode shared by MTU
    union AF { uint4 u[2]; v16h v; };
    AF af[MTU];
    int kb0 = kc * 32 + (hs << 3);     // lane<16: K base 0 ; lane>=16: K base 8
#pragma unroll
    for (int r = 0; r < 2; ++r) {
      int kb = kb0 + (r << 4);         // second run: +16
      int tap = kb >> cin_shift;       // which of the 27 taps (>=27 => K pad)
      int kd = tap / 9;
      int rem = tap - kd * 9;
      int kh = rem / 3;
      int kw = rem - kh * 3;
      bool tap_ok = tap < 27;
      int coff = kb & Cin_mask;
#pragma unroll
      for (int i = 0; i < MTU; ++i) {
        uint4 val = {0u, 0u, 0u, 0u};
        if (tap_ok) {
          int iz = oz[i] * stride + kd - 1;
          int iy = oy[i] * stride + kh - 1;
          int ix = ox[i] * stride + kw - 1;
          if ((unsigned)iz < (unsigned)Din && (unsigned)iy < (unsigned)Hin &&
              (unsigned)ix < (unsigned)Win) {
            size_t off = ((size_t)((iz * Hin + iy) * Win + ix) << cin_shift)
                         + coff;
            val = *(const uint4*)(act + off);   // 8 f16, 16B aligned
          }
        }
        af[i].u[r] = val;
      }
    }

    const _Float16* bchunk = pBl + (size_t)kc * chunk_stride;
    if (kc + 1 < KC)
      __builtin_prefetch(pBl + (size_t)(kc + 1) * chunk_stride, 0, 1);

#pragma unroll
    for (int j = 0; j < NTU; ++j) {
      union { uint4 u[2]; v16h v; } bf;
      const uint4* bp = (const uint4*)(bchunk + (size_t)j * 512);
      bf.u[0] = bp[0];
      bf.u[1] = bp[1];
#pragma unroll
      for (int i = 0; i < MTU; ++i)
        acc[i][j] = __builtin_amdgcn_wmma_f32_16x16x32_f16(
            false, af[i].v, false, bf.v, (short)0, acc[i][j], false, false);
    }
  }

  // ---- epilogue: C/D layout -> lane holds column n, 8 rows in VGPRs
#pragma unroll
  for (int i = 0; i < MTU; ++i) {
    size_t mbase = (size_t)(mtbase + i) * 16 + (hs << 3);
#pragma unroll
    for (int j = 0; j < NTU; ++j) {
      int n = (ntbase + j) * 16 + lr;
      float bv = bias ? bias[n] : 0.f;
#pragma unroll
      for (int r = 0; r < 8; ++r)
        out[(mbase + r) * Cout + n] = acc[i][j][r] + bv;
    }
  }
}

// ---------------------------------------------------------------------------
// Per-voxel GroupNorm (16 groups) + optional residual + ReLU -> f16
// One thread per (voxel, group); group size C/16 in {2,4,8}, channels contig.
// ---------------------------------------------------------------------------
__global__ __launch_bounds__(256) void k_gn_relu_f16(
    const float* __restrict__ xin, const float* __restrict__ gamma,
    const float* __restrict__ beta, const _Float16* __restrict__ res,
    _Float16* __restrict__ out, int C, int gs, long long M)
{
  long long t = (long long)blockIdx.x * 256 + threadIdx.x;
  if (t >= M * 16) return;
  long long vox = t >> 4;
  int g = (int)(t & 15);
  size_t base = (size_t)vox * C + (size_t)g * gs;
  float vals[8];
  float mean = 0.f;
#pragma unroll 8
  for (int i = 0; i < gs; ++i) { vals[i] = xin[base + i]; mean += vals[i]; }
  float inv = 1.f / (float)gs;
  mean *= inv;
  float var = 0.f;
#pragma unroll 8
  for (int i = 0; i < gs; ++i) { float d = vals[i] - mean; var += d * d; }
  var *= inv;
  float rstd = rsqrtf(var + 1e-5f);
  int c0 = g * gs;
#pragma unroll 8
  for (int i = 0; i < gs; ++i) {
    float y = (vals[i] - mean) * rstd * gamma[c0 + i] + beta[c0 + i];
    if (res) y += (float)res[base + i];
    y = fmaxf(y, 0.f);
    out[base + i] = (_Float16)y;
  }
}

// Final GroupNorm + ReLU in fp32 (safe in place: each thread's read set ==
// its write set).
__global__ __launch_bounds__(256) void k_gn_final_f32(
    const float* __restrict__ xin, const float* __restrict__ gamma,
    const float* __restrict__ beta, float* __restrict__ out,
    int C, int gs, long long M)
{
  long long t = (long long)blockIdx.x * 256 + threadIdx.x;
  if (t >= M * 16) return;
  long long vox = t >> 4;
  int g = (int)(t & 15);
  size_t base = (size_t)vox * C + (size_t)g * gs;
  float vals[8];
  float mean = 0.f;
#pragma unroll 8
  for (int i = 0; i < gs; ++i) { vals[i] = xin[base + i]; mean += vals[i]; }
  float inv = 1.f / (float)gs;
  mean *= inv;
  float var = 0.f;
#pragma unroll 8
  for (int i = 0; i < gs; ++i) { float d = vals[i] - mean; var += d * d; }
  var *= inv;
  float rstd = rsqrtf(var + 1e-5f);
  int c0 = g * gs;
#pragma unroll 8
  for (int i = 0; i < gs; ++i) {
    float y = (vals[i] - mean) * rstd * gamma[c0 + i] + beta[c0 + i];
    out[base + i] = fmaxf(y, 0.f);
  }
}

// ---------------------------------------------------------------------------
// Final layout: h [Z=8,Y=64,X=64,C=128] fp32 -> out [C,Z,64,64] with
// transpose->permute(0,1,2,4,3)->rot90(k=1,(3,4)):  out[c,z,i,j] = h[z,63-i,j,c]
// ---------------------------------------------------------------------------
__global__ __launch_bounds__(256) void k_final_permute(
    const float* __restrict__ h, float* __restrict__ out)
{
  int idx = blockIdx.x * 256 + threadIdx.x;   // 128*8*64*64 = 4194304
  int j = idx & 63;
  int i = (idx >> 6) & 63;
  int z = (idx >> 12) & 7;
  int c = idx >> 15;
  out[idx] = h[(((size_t)z * 64 + (63 - i)) * 64 + j) * 128 + c];
}

// ---------------------------------------------------------------------------
extern "C" void kernel_launch(void* const* d_in, const int* in_sizes, int n_in,
                              void* d_out, int out_size, void* d_ws, size_t ws_size,
                              hipStream_t stream)
{
  (void)in_sizes; (void)n_in; (void)out_size; (void)ws_size;

  const float* vf    = (const float*)d_in[0];
  const int*   coors = (const int*)  d_in[1];
  const float* b_in  = (const float*)d_in[3];
  const float* g_in  = (const float*)d_in[4];  const float* be_in  = (const float*)d_in[5];
  const float* g_c1  = (const float*)d_in[7];  const float* be_c1  = (const float*)d_in[8];
  const float* g_r1a1= (const float*)d_in[10]; const float* be_r1a1= (const float*)d_in[11];
  const float* g_r1a2= (const float*)d_in[13]; const float* be_r1a2= (const float*)d_in[14];
  const float* g_r1b1= (const float*)d_in[16]; const float* be_r1b1= (const float*)d_in[17];
  const float* g_r1b2= (const float*)d_in[19]; const float* be_r1b2= (const float*)d_in[20];
  const float* g_c2  = (const float*)d_in[22]; const float* be_c2  = (const float*)d_in[23];
  const float* g_r2a1= (const float*)d_in[25]; const float* be_r2a1= (const float*)d_in[26];
  const float* g_r2a2= (const float*)d_in[28]; const float* be_r2a2= (const float*)d_in[29];
  const float* g_r2b1= (const float*)d_in[31]; const float* be_r2b1= (const float*)d_in[32];
  const float* g_r2b2= (const float*)d_in[34]; const float* be_r2b2= (const float*)d_in[35];
  const float* b_out = (const float*)d_in[37];
  const float* g_out = (const float*)d_in[38]; const float* be_out = (const float*)d_in[39];

  char* ws = (char*)d_ws;
  const size_t MB = 1024ull * 1024ull;
  _Float16* pW   = (_Float16*)(ws);              //   8 MB packed weights
  float*    acc  = (float*)   (ws +   8 * MB);   // 256 MB fp32 conv accum
  _Float16* actA = (_Float16*)(ws + 264 * MB);   // 128 MB f16 activations
  _Float16* actB = (_Float16*)(ws + 392 * MB);   // 128 MB f16 activations

  // -------- pack all 12 conv weights into WMMA B-fragment order ----------
  const int widx[12] = {2, 6, 9, 12, 15, 18, 21, 24, 27, 30, 33, 36};
  const int wcin[12] = {16, 32, 64, 64, 64, 64, 64, 128, 128, 128, 128, 128};
  const int wcout[12]= {32, 64, 64, 64, 64, 64, 128, 128, 128, 128, 128, 128};
  _Float16* wp[12];
  size_t off = 0;
  for (int l = 0; l < 12; ++l) {
    int Ktot = 27 * wcin[l];
    int KC   = (Ktot + 31) / 32;
    int Nt   = wcout[l] / 16;
    long long total = (long long)Nt * KC * 512;
    wp[l] = pW + off;
    k_pack<<<(unsigned)((total + 255) / 256), 256, 0, stream>>>(
        (const float*)d_in[widx[l]], wp[l], wcout[l], Ktot, Nt, total);
    off += (size_t)total;
  }

  // -------- scatter input (identity fill) to dense f16 grid --------------
  const int NVOX = Z_DIM * Y_DIM * X_DIM;   // 2,097,152
  k_scatter<<<(NVOX + 255) / 256, 256, 0, stream>>>(vf, coors, actB, NVOX);

  // -------- helpers -------------------------------------------------------
  auto conv = [&](const _Float16* in, int Din, int Hin, int Win, int cin_shift,
                  int Dout, int Hout, int Wout, int Cout, int stride,
                  const _Float16* pB, const float* bias) {
    long long M = (long long)Dout * Hout * Wout;
    int Nt = Cout / 16;
    int Mt = (int)(M / 16);
    int KC = ((27 << cin_shift) + 31) / 32;
    int ho_shift = __builtin_ctz(Hout);
    int wo_shift = __builtin_ctz(Wout);
    if (Nt == 2) {
      dim3 grid(Mt / 16, 1);              // MTU=2 -> 16 m-tiles per block
      k_conv_wmma<2, 2><<<grid, 256, 0, stream>>>(
          in, pB, bias, acc, Din, Hin, Win, cin_shift,
          Hout - 1, Wout - 1, ho_shift, wo_shift, Cout, stride, Nt, KC);
    } else {
      dim3 grid(Mt / 16, Nt / 4);
      k_conv_wmma<2, 4><<<grid, 256, 0, stream>>>(
          in, pB, bias, acc, Din, Hin, Win, cin_shift,
          Hout - 1, Wout - 1, ho_shift, wo_shift, Cout, stride, Nt, KC);
    }
  };
  auto gn = [&](const float* g, const float* b, const _Float16* res,
                _Float16* o, int C, long long M) {
    long long total = M * 16;
    k_gn_relu_f16<<<(unsigned)((total + 255) / 256), 256, 0, stream>>>(
        acc, g, b, res, o, C, C / 16, M);
  };

  const long long M0 = (long long)NVOX;     // 32x256x256
  const long long M1 = 16LL * 128 * 128;    // 262,144
  const long long M2 = 8LL * 64 * 64;       // 32,768

  // -------- network -------------------------------------------------------
  // conv_in + b_in, GN, ReLU
  conv(actB, 32, 256, 256, 4,  32, 256, 256, 32, 1, wp[0], b_in);
  gn(g_in, be_in, nullptr, actA, 32, M0);
  // c1 (stride 2), GN, ReLU
  conv(actA, 32, 256, 256, 5,  16, 128, 128, 64, 2, wp[1], nullptr);
  gn(g_c1, be_c1, nullptr, actB, 64, M1);
  // r1a: x = actB
  conv(actB, 16, 128, 128, 6,  16, 128, 128, 64, 1, wp[2], nullptr);
  gn(g_r1a1, be_r1a1, nullptr, actA, 64, M1);
  conv(actA, 16, 128, 128, 6,  16, 128, 128, 64, 1, wp[3], nullptr);
  gn(g_r1a2, be_r1a2, actB, actA, 64, M1);            // + residual(actB), out actA
  // r1b: x = actA
  conv(actA, 16, 128, 128, 6,  16, 128, 128, 64, 1, wp[4], nullptr);
  gn(g_r1b1, be_r1b1, nullptr, actB, 64, M1);
  conv(actB, 16, 128, 128, 6,  16, 128, 128, 64, 1, wp[5], nullptr);
  gn(g_r1b2, be_r1b2, actA, actB, 64, M1);            // out actB
  // c2 (stride 2), GN, ReLU
  conv(actB, 16, 128, 128, 6,  8, 64, 64, 128, 2, wp[6], nullptr);
  gn(g_c2, be_c2, nullptr, actA, 128, M2);
  // r2a: x = actA
  conv(actA, 8, 64, 64, 7,  8, 64, 64, 128, 1, wp[7], nullptr);
  gn(g_r2a1, be_r2a1, nullptr, actB, 128, M2);
  conv(actB, 8, 64, 64, 7,  8, 64, 64, 128, 1, wp[8], nullptr);
  gn(g_r2a2, be_r2a2, actA, actB, 128, M2);           // out actB
  // r2b: x = actB
  conv(actB, 8, 64, 64, 7,  8, 64, 64, 128, 1, wp[9], nullptr);
  gn(g_r2b1, be_r2b1, nullptr, actA, 128, M2);
  conv(actA, 8, 64, 64, 7,  8, 64, 64, 128, 1, wp[10], nullptr);
  gn(g_r2b2, be_r2b2, actB, actA, 128, M2);           // out actA
  // conv_out + b_out, GN, ReLU (fp32, in place on acc)
  conv(actA, 8, 64, 64, 7,  8, 64, 64, 128, 1, wp[11], b_out);
  {
    long long total = M2 * 16;
    k_gn_final_f32<<<(unsigned)((total + 255) / 256), 256, 0, stream>>>(
        acc, g_out, be_out, acc, 128, 8, M2);
  }
  // transpose -> permute -> rot90 gather into d_out
  k_final_permute<<<(128 * 8 * 64 * 64) / 256, 256, 0, stream>>>(acc, (float*)d_out);
}
